// LinearAttention_expand_rel_SAM_Swin_75041668596126
// MI455X (gfx1250) — matrix operand order repaired and checked by reading, not compiled, not run on hardware
//
#include <hip/hip_runtime.h>
#include <hip/hip_bf16.h>

typedef _Float16 half_t;
typedef __attribute__((ext_vector_type(16))) _Float16 v16h;
typedef __attribute__((ext_vector_type(8)))  _Float16 v8h;
typedef __attribute__((ext_vector_type(8)))  float    v8f;

#define DIMC   256
#define HEADS  4
#define NQ     4096
#define NK     256
#define HW     4096
#define SCALE  0.125f

// ---------------- WMMA fragment helpers (CDNA5 16x16x32 f16) ----------------
// A (16x32, MxK): lane<16 -> M=lane, K = {0..7,16..23}; lane>=16 -> K={8..15,24..31}
__device__ __forceinline__ v16h load_a_frag(const half_t* __restrict__ A, int lda,
                                            int m0, int k0, int lane) {
  int row = m0 + (lane & 15);
  int kh  = lane >> 4;
  const half_t* p = A + (size_t)row * lda + k0 + kh * 8;
  v8h lo = *(const v8h*)p;         // K = k0+kh*8 .. +7
  v8h hi = *(const v8h*)(p + 16);  // K = k0+16+kh*8 .. +7
  v16h r;
#pragma unroll
  for (int i = 0; i < 8; ++i) { r[i] = lo[i]; r[i + 8] = hi[i]; }
  return r;
}

// B (32x16, KxN) supplied as Bt row-major [N][K]: lane holds col n, K = {k0+kh*8..+7, k0+16+kh*8..+7}
__device__ __forceinline__ v16h load_b_frag(const half_t* __restrict__ Bt, int ldb,
                                            int n0, int k0, int lane) {
  int col = n0 + (lane & 15);
  int kh  = lane >> 4;
  const half_t* p = Bt + (size_t)col * ldb + k0 + kh * 8;
  v8h lo = *(const v8h*)p;
  v8h hi = *(const v8h*)(p + 16);
  v16h r;
#pragma unroll
  for (int i = 0; i < 8; ++i) { r[i] = lo[i]; r[i + 8] = hi[i]; }
  return r;
}

__device__ __forceinline__ v8f wmma16(v16h a, v16h b, v8f c) {
  return __builtin_amdgcn_wmma_f32_16x16x32_f16(false, a, false, b, (short)0, c, false, false);
}

// Async global->LDS copy of one 16B chunk per lane (CDNA5, ASYNCcnt-tracked).
__device__ __forceinline__ void async_g2l_b128(unsigned lds_off, const void* gptr) {
  unsigned long long ga = (unsigned long long)(uintptr_t)gptr;
  asm volatile("global_load_async_to_lds_b128 %0, %1, off" :: "v"(lds_off), "v"(ga) : "memory");
}
__device__ __forceinline__ void wait_async0() {
  asm volatile("s_wait_asynccnt 0x0" ::: "memory");
}

// ---------------- small prep kernels ----------------
__global__ void k_f32_to_f16(const float* __restrict__ src, half_t* __restrict__ dst, int n) {
  int i = blockIdx.x * blockDim.x + threadIdx.x;
  if (i < n) dst[i] = (half_t)src[i];
}

// biasB[h][p] = rel_table[rel_index_flat[p]][h],  p in [0, 1024*1024)
__global__ void k_bias(const float* __restrict__ table, const int* __restrict__ ridx,
                       half_t* __restrict__ biasB) {
  int t = blockIdx.x * blockDim.x + threadIdx.x;   // 4 * 2^20 threads
  int p = t & ((1 << 20) - 1);
  int h = t >> 20;
  biasB[t] = (half_t)table[ridx[p] * HEADS + h];
}

// depthwise 3x3 pad1 on f32 NCHW input -> f16 [B][HW][C] (K-contiguous GEMM-A layout)
__global__ void k_dwconv_in(const float* __restrict__ x, const float* __restrict__ w,
                            half_t* __restrict__ A1) {
  int t   = blockIdx.x * blockDim.x + threadIdx.x;  // B*HW*256
  int c   = t & 255;
  int pos = (t >> 8) & (HW - 1);
  int b   = t >> 20;
  int y = pos >> 6, xx = pos & 63;
  const float* xp = x + (size_t)(b * DIMC + c) * HW;
  const float* wp = w + c * 9;
  float acc = 0.f;
#pragma unroll
  for (int dy = 0; dy < 3; ++dy) {
    int yy = y + dy - 1;
    if (yy < 0 || yy > 63) continue;
#pragma unroll
    for (int dx = 0; dx < 3; ++dx) {
      int xw = xx + dx - 1;
      if (xw < 0 || xw > 63) continue;
      acc += xp[(yy << 6) + xw] * wp[dy * 3 + dx];
    }
  }
  A1[t] = (half_t)acc;
}

// ---------------- GEMM 1: [65536x256] x [256x768] -> q/k/v (scatter epilogue) ----------------
// Block = 4 waves sharing one 16-row A tile; A staged once via async global->LDS.
__global__ void k_gemm_qkv(const half_t* __restrict__ A1, const half_t* __restrict__ Wq,
                           half_t* __restrict__ qA, half_t* __restrict__ kf,
                           half_t* __restrict__ vf) {
  __shared__ half_t As[16 * 256];                    // 8 KB, contiguous tile
  int lane = threadIdx.x & 31;
  int wave = threadIdx.x >> 5;
  int m0 = blockIdx.x * 16;
  int n0 = (blockIdx.y * 4 + wave) * 64;

  unsigned ldsBase = (unsigned)(uintptr_t)&As[0];
  const char* gsrc = (const char*)(A1 + (size_t)m0 * 256);
#pragma unroll
  for (int i = 0; i < 4; ++i)                        // 128 thr * 4 * 16B = 8 KB
    async_g2l_b128(ldsBase + threadIdx.x * 16 + i * 2048,
                   gsrc + threadIdx.x * 16 + i * 2048);
  wait_async0();
  __syncthreads();

  v8f acc[4] = {};
  for (int k0 = 0; k0 < 256; k0 += 32) {
    v16h a = load_a_frag(As, 256, 0, k0, lane);      // ds_load path
#pragma unroll
    for (int t = 0; t < 4; ++t) {
      v16h b = load_b_frag(Wq, 256, n0 + t * 16, k0, lane);
      acc[t] = wmma16(a, b, acc[t]);
    }
  }
  int nlane = lane & 15;
  int mofs  = (lane >> 4) * 8;
#pragma unroll
  for (int t = 0; t < 4; ++t) {
    int n = n0 + t * 16 + nlane;
#pragma unroll
    for (int r = 0; r < 8; ++r) {
      int m = m0 + mofs + r;
      int b = m >> 12, pos = m & 4095;
      float v = acc[t][r];
      if (n < 256) {                       // q -> [b,head,pos,dh], scale folded in
        int head = n & 3, dh = n >> 2;
        qA[(((size_t)(b * 4 + head) * NQ + pos) << 6) + dh] = (half_t)(v * SCALE);
      } else if (n < 512) {                // k -> [b,c,pos] for pooling
        kf[((size_t)(b * DIMC + (n - 256)) << 12) + pos] = (half_t)v;
      } else {                             // v -> [b,c,pos]
        vf[((size_t)(b * DIMC + (n - 512)) << 12) + pos] = (half_t)v;
      }
    }
  }
}

// 4x4 max-pool + head split: kB = [bh][j][dh] (Bt for S-gemm), vB = [bh][dh][j] (Bt for PV)
__global__ void k_pool_split(const half_t* __restrict__ kf, const half_t* __restrict__ vf,
                             half_t* __restrict__ kB, half_t* __restrict__ vB) {
  int t  = blockIdx.x * blockDim.x + threadIdx.x;  // 16*4*256*64
  int dh = t & 63;
  int j  = (t >> 6) & 255;
  int h  = (t >> 14) & 3;
  int b  = t >> 16;
  int ch = dh * 4 + h;
  int jr = j >> 4, jc = j & 15;
  const half_t* kp = kf + ((size_t)(b * DIMC + ch) << 12);
  const half_t* vp = vf + ((size_t)(b * DIMC + ch) << 12);
  float mk = -1e30f, mv = -1e30f;
#pragma unroll
  for (int dy = 0; dy < 4; ++dy)
#pragma unroll
    for (int dx = 0; dx < 4; ++dx) {
      int pos = ((jr * 4 + dy) << 6) + jc * 4 + dx;
      mk = fmaxf(mk, (float)kp[pos]);
      mv = fmaxf(mv, (float)vp[pos]);
    }
  int bh = b * 4 + h;
  kB[(((size_t)bh * NK + j) << 6) + dh]   = (half_t)mk;
  vB[(((size_t)bh << 6) + dh) * NK + j]   = (half_t)mv;
}

// ---------------- GEMM 2: S = q.kT + bias -> f32 logits (in d_out attn slice) ----------------
__global__ void k_attn_scores(const half_t* __restrict__ qA, const half_t* __restrict__ kB,
                              const half_t* __restrict__ biasB, float* __restrict__ logits) {
  __shared__ half_t As[16 * 64];                     // 2 KB q tile, contiguous
  int lane = threadIdx.x & 31;
  int wave = threadIdx.x >> 5;
  int bh = blockIdx.y;
  int h  = bh & 3;
  int i0 = blockIdx.x * 16;
  int n0 = wave * 64;
  const half_t* Ap = qA + ((size_t)bh * NQ) * 64;
  const half_t* Bp = kB + ((size_t)bh * NK) * 64;

  unsigned ldsBase = (unsigned)(uintptr_t)&As[0];
  const char* gsrc = (const char*)(Ap + (size_t)i0 * 64);
  async_g2l_b128(ldsBase + threadIdx.x * 16, gsrc + threadIdx.x * 16);  // 128*16B = 2 KB
  wait_async0();
  __syncthreads();

  v8f acc[4] = {};
  for (int k0 = 0; k0 < 64; k0 += 32) {
    v16h a = load_a_frag(As, 64, 0, k0, lane);
#pragma unroll
    for (int t = 0; t < 4; ++t) {
      v16h b = load_b_frag(Bp, 64, n0 + t * 16, k0, lane);
      acc[t] = wmma16(a, b, acc[t]);
    }
  }
  int nlane = lane & 15;
  int mofs  = (lane >> 4) * 8;
#pragma unroll
  for (int t = 0; t < 4; ++t) {
    int j = n0 + t * 16 + nlane;
#pragma unroll
    for (int r = 0; r < 8; ++r) {
      int i = i0 + mofs + r;
      float bias = (float)biasB[((size_t)h * NQ + i) * NK + j];
      logits[((size_t)bh * NQ + i) * NK + j] = acc[t][r] + bias;
    }
  }
}

// wave-per-row softmax over 256 cols; writes f32 attn (in place) + f16 P
__global__ void k_softmax(float* __restrict__ attn, half_t* __restrict__ P) {
  int lane = threadIdx.x & 31;
  int wave = threadIdx.x >> 5;
  size_t row = (size_t)blockIdx.x * 8 + wave;
  float* rp = attn + row * NK;
  float v[8];
  float mx = -1e30f;
#pragma unroll
  for (int t = 0; t < 8; ++t) { v[t] = rp[lane + t * 32]; mx = fmaxf(mx, v[t]); }
#pragma unroll
  for (int s = 16; s >= 1; s >>= 1) mx = fmaxf(mx, __shfl_xor(mx, s, 32));
  float sum = 0.f;
#pragma unroll
  for (int t = 0; t < 8; ++t) { v[t] = __expf(v[t] - mx); sum += v[t]; }
#pragma unroll
  for (int s = 16; s >= 1; s >>= 1) sum += __shfl_xor(sum, s, 32);
  float inv = 1.f / sum;
  half_t* pp = P + row * NK;
#pragma unroll
  for (int t = 0; t < 8; ++t) {
    float o = v[t] * inv;
    rp[lane + t * 32] = o;
    pp[lane + t * 32] = (half_t)o;
  }
}

// ---------------- GEMM 3: out = P.v -> oA [B][pos][c] f16 ----------------
__global__ void k_attn_out(const half_t* __restrict__ P, const half_t* __restrict__ vB,
                           half_t* __restrict__ oA) {
  int lane = threadIdx.x & 31;
  int wave = threadIdx.x >> 5;
  int bh = blockIdx.y;
  int b = bh >> 2, h = bh & 3;
  int i0 = (blockIdx.x * 4 + wave) * 16;
  const half_t* Ap = P  + ((size_t)bh * NQ) * NK;
  const half_t* Bp = vB + ((size_t)bh * 64) * NK;
  v8f acc[4] = {};
  for (int k0 = 0; k0 < 256; k0 += 32) {
    v16h a = load_a_frag(Ap, 256, i0, k0, lane);
#pragma unroll
    for (int t = 0; t < 4; ++t) {
      v16h bf = load_b_frag(Bp, 256, t * 16, k0, lane);
      acc[t] = wmma16(a, bf, acc[t]);
    }
  }
  int nlane = lane & 15;
  int mofs  = (lane >> 4) * 8;
#pragma unroll
  for (int t = 0; t < 4; ++t) {
    int dh = t * 16 + nlane;
    int c  = dh * 4 + h;
#pragma unroll
    for (int r = 0; r < 8; ++r) {
      int i = i0 + mofs + r;
      oA[(((size_t)b * HW + i) << 8) + c] = (half_t)acc[t][r];
    }
  }
}

// depthwise 3x3 on [B][HW][C] f16 layout (both in & out K-contiguous)
__global__ void k_dwconv_out(const half_t* __restrict__ oA, const float* __restrict__ w,
                             half_t* __restrict__ dw2) {
  int t   = blockIdx.x * blockDim.x + threadIdx.x;
  int c   = t & 255;
  int pos = (t >> 8) & 4095;
  int b   = t >> 20;
  int y = pos >> 6, xx = pos & 63;
  const half_t* ip = oA + ((size_t)b << 20);
  const float*  wp = w + c * 9;
  float acc = 0.f;
#pragma unroll
  for (int dy = 0; dy < 3; ++dy) {
    int yy = y + dy - 1;
    if (yy < 0 || yy > 63) continue;
#pragma unroll
    for (int dx = 0; dx < 3; ++dx) {
      int xw = xx + dx - 1;
      if (xw < 0 || xw > 63) continue;
      acc += (float)ip[(((yy << 6) + xw) << 8) + c] * wp[dy * 3 + dx];
    }
  }
  dw2[t] = (half_t)acc;
}

// ---------------- GEMM 4: [65536x256] x [256x256] -> f32 NCHW into d_out ----------------
__global__ void k_gemm_out(const half_t* __restrict__ A, const half_t* __restrict__ Wo,
                           float* __restrict__ out) {
  __shared__ half_t As[16 * 256];                    // 8 KB shared A tile
  int lane = threadIdx.x & 31;
  int wave = threadIdx.x >> 5;
  int m0 = blockIdx.x * 16;
  int n0 = wave * 64;

  unsigned ldsBase = (unsigned)(uintptr_t)&As[0];
  const char* gsrc = (const char*)(A + (size_t)m0 * 256);
#pragma unroll
  for (int i = 0; i < 4; ++i)
    async_g2l_b128(ldsBase + threadIdx.x * 16 + i * 2048,
                   gsrc + threadIdx.x * 16 + i * 2048);
  wait_async0();
  __syncthreads();

  v8f acc[4] = {};
  for (int k0 = 0; k0 < 256; k0 += 32) {
    v16h a = load_a_frag(As, 256, 0, k0, lane);
#pragma unroll
    for (int t = 0; t < 4; ++t) {
      v16h bf = load_b_frag(Wo, 256, n0 + t * 16, k0, lane);
      acc[t] = wmma16(a, bf, acc[t]);
    }
  }
  int nlane = lane & 15;
  int mofs  = (lane >> 4) * 8;
#pragma unroll
  for (int t = 0; t < 4; ++t) {
    int n = n0 + t * 16 + nlane;
#pragma unroll
    for (int r = 0; r < 8; ++r) {
      int m = m0 + mofs + r;
      int b = m >> 12, pos = m & 4095;
      out[((size_t)(b * DIMC + n) << 12) + pos] = acc[t][r];
    }
  }
}

extern "C" void kernel_launch(void* const* d_in, const int* in_sizes, int n_in,
                              void* d_out, int out_size, void* d_ws, size_t ws_size,
                              hipStream_t stream) {
  const float* x     = (const float*)d_in[0];
  const float* dwq_w = (const float*)d_in[1];
  const float* pwq_w = (const float*)d_in[2];
  const float* dwo_w = (const float*)d_in[3];
  const float* pwo_w = (const float*)d_in[4];
  const float* rtab  = (const float*)d_in[5];
  const int*   ridx  = (const int*)d_in[6];

  char* ws = (char*)d_ws;
  size_t off = 0;
  auto alloc = [&](size_t bytes) -> void* {
    void* p = ws + off; off += (bytes + 255) & ~(size_t)255; return p;
  };
  half_t* wQKV  = (half_t*)alloc((size_t)768 * 256 * 2);
  half_t* wOUT  = (half_t*)alloc((size_t)256 * 256 * 2);
  half_t* biasB = (half_t*)alloc((size_t)4 * 1048576 * 2);
  half_t* A1    = (half_t*)alloc((size_t)16777216 * 2);  // reused as P after GEMM1
  half_t* qA    = (half_t*)alloc((size_t)16777216 * 2);
  half_t* kf    = (half_t*)alloc((size_t)16777216 * 2);  // reused as oA after pool
  half_t* vf    = (half_t*)alloc((size_t)16777216 * 2);  // reused as dw2 after pool
  half_t* kB    = (half_t*)alloc((size_t)1048576 * 2);
  half_t* vB    = (half_t*)alloc((size_t)1048576 * 2);
  half_t* P = A1;  half_t* oA = kf;  half_t* dw2 = vf;

  float* out_main = (float*)d_out;
  float* attn     = out_main + (size_t)16777216;  // [16,4,4096,256] f32

  k_f32_to_f16 <<<768,   256, 0, stream>>>(pwq_w, wQKV, 768 * 256);
  k_f32_to_f16 <<<256,   256, 0, stream>>>(pwo_w, wOUT, 256 * 256);
  k_bias       <<<16384, 256, 0, stream>>>(rtab, ridx, biasB);
  k_dwconv_in  <<<65536, 256, 0, stream>>>(x, dwq_w, A1);
  k_gemm_qkv   <<<dim3(4096, 3), 128, 0, stream>>>(A1, wQKV, qA, kf, vf);
  k_pool_split <<<4096,  256, 0, stream>>>(kf, vf, kB, vB);
  k_attn_scores<<<dim3(256, 64), 128, 0, stream>>>(qA, kB, biasB, attn);
  k_softmax    <<<32768, 256, 0, stream>>>(attn, P);
  k_attn_out   <<<dim3(64, 64), 128, 0, stream>>>(P, vB, oA);
  k_dwconv_out <<<65536, 256, 0, stream>>>(oA, dwo_w, dw2);
  k_gemm_out   <<<4096,  128, 0, stream>>>(dw2, wOUT, out_main);
}
